// GLMBlock_67396626809291
// MI455X (gfx1250) — compile-verified, matrix operand order checked by reading, not run on hardware
//
#include <hip/hip_runtime.h>
#include <cstdint>
#include <cstddef>

// ---------------------------------------------------------------- types
typedef __bf16 bf16;
typedef __attribute__((ext_vector_type(16))) __bf16 v16bf;
typedef __attribute__((ext_vector_type(8)))  __bf16 v8bf;
typedef __attribute__((ext_vector_type(8)))  float  v8f;

#define WMMA_BF16(a, b, c) \
  __builtin_amdgcn_wmma_f32_16x16x32_bf16(false, (a), false, (b), (short)0, (c), false, false)

// ---------------------------------------------------------------- problem constants
static constexpr int kB    = 4;
static constexpr int kS    = 1024;
static constexpr int kH    = 4096;
static constexpr int kNH   = 32;
static constexpr int kNKV  = 2;
static constexpr int kHD   = 128;
static constexpr int kG    = kNH / kNKV;      // 16
static constexpr int kFFN  = 13696;
static constexpr int kNQKV = (kNH + 2 * kNKV) * kHD;  // 4608
static constexpr int kM    = kB * kS;         // 4096 tokens
static constexpr float kEPS   = 1e-5f;
static constexpr float kSCALE = 0.08838834764831845f; // 1/sqrt(128)

// ---------------------------------------------------------------- helpers
__device__ inline v8f zero8() {
  v8f z;
#pragma unroll
  for (int i = 0; i < 8; ++i) z[i] = 0.0f;
  return z;
}

// Load one 16-bit WMMA fragment slice for this lane:
// elements 0..7 from p[0..7], elements 8..15 from p[16..23]
// (matches CDNA5 16-bit A/B 16x32 VGPR layout after the caller picks
//  row = lane%16 and kbase = (lane/16)*8).
__device__ inline v16bf load_frag(const bf16* p) {
  union { v16bf v; v8bf h[2]; } u;
  u.h[0] = *(const v8bf*)(p);
  u.h[1] = *(const v8bf*)(p + 16);
  return u.v;
}

__device__ inline float rmax16(float v) {
  v = fmaxf(v, __shfl_xor(v, 1, 32));
  v = fmaxf(v, __shfl_xor(v, 2, 32));
  v = fmaxf(v, __shfl_xor(v, 4, 32));
  v = fmaxf(v, __shfl_xor(v, 8, 32));
  return v;
}
__device__ inline float rsum16(float v) {
  v += __shfl_xor(v, 1, 32);
  v += __shfl_xor(v, 2, 32);
  v += __shfl_xor(v, 4, 32);
  v += __shfl_xor(v, 8, 32);
  return v;
}

// ---------------------------------------------------------------- elementwise kernels
__global__ void cast_f32_bf16(const float* __restrict__ in, bf16* __restrict__ out, size_t n) {
  size_t i = (size_t)blockIdx.x * blockDim.x + threadIdx.x;
  size_t stride = (size_t)gridDim.x * blockDim.x;
  for (; i < n; i += stride) out[i] = (bf16)in[i];
}

__global__ void rmsnorm_kernel(const float* __restrict__ x, const float* __restrict__ w,
                               bf16* __restrict__ out, int H) {
  int row = blockIdx.x;
  const float* xr = x + (size_t)row * H;
  float ss = 0.0f;
  for (int i = threadIdx.x; i < H; i += blockDim.x) { float v = xr[i]; ss += v * v; }
#pragma unroll
  for (int m = 16; m >= 1; m >>= 1) ss += __shfl_xor(ss, m, 32);
  __shared__ float red[8];
  if ((threadIdx.x & 31) == 0) red[threadIdx.x >> 5] = ss;
  __syncthreads();
  float tot = 0.0f;
#pragma unroll
  for (int i = 0; i < 8; ++i) tot += red[i];
  float scale = rsqrtf(tot / (float)H + kEPS);
  bf16* orow = out + (size_t)row * H;
  for (int i = threadIdx.x; i < H; i += blockDim.x)
    orow[i] = (bf16)(xr[i] * scale * w[i]);
}

// RoPE + layout change: qkv fp32 [tok, 4608] -> q[b,NH,s,HD], k/v[b,NKV,s,HD] bf16
__global__ void rope_kernel(const float* __restrict__ qkv, const int* __restrict__ pos,
                            bf16* __restrict__ q, bf16* __restrict__ k, bf16* __restrict__ v) {
  int tok = blockIdx.x;
  int b = tok / kS, s = tok % kS;
  const float* row = qkv + (size_t)tok * kNQKV;
  float p = (float)pos[tok];
  int t = threadIdx.x;
  // rotary halves (dims 0..63, 32 pairs) for NH q heads + NKV k heads
  for (int idx = t; idx < (kNH + kNKV) * 32; idx += blockDim.x) {
    int h = idx >> 5, pr = idx & 31;
    float inv = __powf(10000.0f, -(2.0f * (float)pr) / 64.0f);
    float ang = p * inv;
    float c = __cosf(ang), sn = __sinf(ang);
    int off = h * kHD;
    float x1 = row[off + 2 * pr], x2 = row[off + 2 * pr + 1];
    float o1 = x1 * c - x2 * sn;
    float o2 = x1 * sn + x2 * c;
    if (h < kNH) {
      size_t base = ((size_t)(b * kNH + h) * kS + s) * kHD;
      q[base + 2 * pr] = (bf16)o1; q[base + 2 * pr + 1] = (bf16)o2;
    } else {
      size_t base = ((size_t)(b * kNKV + (h - kNH)) * kS + s) * kHD;
      k[base + 2 * pr] = (bf16)o1; k[base + 2 * pr + 1] = (bf16)o2;
    }
  }
  // pass-through dims 64..127 of q and k
  for (int idx = t; idx < (kNH + kNKV) * 64; idx += blockDim.x) {
    int h = idx >> 6, d = 64 + (idx & 63);
    float x = row[h * kHD + d];
    if (h < kNH)
      q[((size_t)(b * kNH + h) * kS + s) * kHD + d] = (bf16)x;
    else
      k[((size_t)(b * kNKV + (h - kNH)) * kS + s) * kHD + d] = (bf16)x;
  }
  // v copy
  for (int idx = t; idx < kNKV * kHD; idx += blockDim.x) {
    int h = idx >> 7, d = idx & 127;
    v[((size_t)(b * kNKV + h) * kS + s) * kHD + d] =
        (bf16)row[(kNH + kNKV) * kHD + h * kHD + d];
  }
}

// ---------------------------------------------------------------- generic WMMA GEMM
// C[M,N] = A[M,K](bf16,rowmajor) x B[K,N](bf16,rowmajor) (+bias[N]) (+res[M,N] f32)
// BM=BN=128, BK=32; 8 waves in 2x4 grid, each wave does a 64x32 tile (4x2 wmma tiles).
template <bool OUT_BF16, bool HAS_BIAS, bool HAS_RES>
__global__ __launch_bounds__(256) void gemm_bf16(
    const bf16* __restrict__ A, const bf16* __restrict__ B,
    const float* __restrict__ bias, const float* __restrict__ res,
    void* __restrict__ Cout, int M, int N, int K) {
  constexpr int BM = 128, BN = 128, BK = 32, LDT = BK + 8; // 40 elems / 80B rows
  __shared__ __align__(16) bf16 As[BM * LDT];
  __shared__ __align__(16) bf16 Bs[BN * LDT]; // transposed: [n][k]
  const int tid = threadIdx.x;
  const int wave = tid >> 5, lane = tid & 31;
  const int wrow = wave >> 2, wcol = wave & 3;     // 2 x 4 waves
  const int m0 = blockIdx.y * BM, n0 = blockIdx.x * BN;
  const int wm = wrow * 64, wn = wcol * 32;
  const int lmod = lane & 15, lhalf = lane >> 4;
  const int kb8 = lhalf * 8;

  v8f acc[4][2];
#pragma unroll
  for (int i = 0; i < 4; ++i)
#pragma unroll
    for (int j = 0; j < 2; ++j) acc[i][j] = zero8();

  for (int kb = 0; kb < K; kb += BK) {
    __syncthreads();
    // stage A tile (128x32): 512 chunks of 8 bf16
#pragma unroll
    for (int it = 0; it < 2; ++it) {
      int c = tid + it * 256;
      int r = c >> 2, o = (c & 3) * 8;
      *(v8bf*)&As[r * LDT + o] = *(const v8bf*)&A[(size_t)(m0 + r) * K + kb + o];
    }
    // stage B tile transposed (Bs[n][k]): read rows of B, scatter
#pragma unroll
    for (int it = 0; it < 2; ++it) {
      int c = tid + it * 256;
      int kk = c >> 4, nn = (c & 15) * 8;
      v8bf tv = *(const v8bf*)&B[(size_t)(kb + kk) * N + n0 + nn];
#pragma unroll
      for (int e = 0; e < 8; ++e) Bs[(nn + e) * LDT + kk] = tv[e];
    }
    if (kb + BK < K) __builtin_prefetch(&A[(size_t)m0 * K + kb + BK], 0, 0);
    __syncthreads();

    v16bf af[4], bfr[2];
#pragma unroll
    for (int i = 0; i < 4; ++i)
      af[i] = load_frag(&As[(wm + i * 16 + lmod) * LDT + kb8]);
#pragma unroll
    for (int j = 0; j < 2; ++j)
      bfr[j] = load_frag(&Bs[(wn + j * 16 + lmod) * LDT + kb8]);
#pragma unroll
    for (int i = 0; i < 4; ++i)
#pragma unroll
      for (int j = 0; j < 2; ++j)
        acc[i][j] = WMMA_BF16(af[i], bfr[j], acc[i][j]);
  }

  // epilogue (C layout: row = r + 8*lhalf, col = lmod within each 16x16 tile)
#pragma unroll
  for (int i = 0; i < 4; ++i) {
#pragma unroll
    for (int j = 0; j < 2; ++j) {
      int col = n0 + wn + j * 16 + lmod;
      float bv = HAS_BIAS ? bias[col] : 0.0f;
#pragma unroll
      for (int r = 0; r < 8; ++r) {
        int row = m0 + wm + i * 16 + r + 8 * lhalf;
        float v = acc[i][j][r] + bv;
        if (HAS_RES) v += res[(size_t)row * N + col];
        if constexpr (OUT_BF16)
          ((bf16*)Cout)[(size_t)row * N + col] = (bf16)v;
        else
          ((float*)Cout)[(size_t)row * N + col] = v;
      }
    }
  }
}

// ---------------------------------------------------------------- fused gate/up GEMM
// act[M,FFN] = silu(A@Wg) * (A@Wu), Wg = B[:, :FFN], Wu = B[:, FFN:], B=[K, 2*FFN]
// BM=128, BN=64, BK=32; 8 waves in 4x2, each wave 32x32 (2x2 tiles, two accum sets).
__global__ __launch_bounds__(256) void gemm_gateup(
    const bf16* __restrict__ A, const bf16* __restrict__ B,
    bf16* __restrict__ act, int M, int K) {
  constexpr int BM = 128, BN = 64, BK = 32, LDT = BK + 8;
  constexpr int NTOT = 2 * kFFN;
  __shared__ __align__(16) bf16 As[BM * LDT];
  __shared__ __align__(16) bf16 Bg[BN * LDT];
  __shared__ __align__(16) bf16 Bu[BN * LDT];
  const int tid = threadIdx.x;
  const int wave = tid >> 5, lane = tid & 31;
  const int wrow = wave >> 1, wcol = wave & 1;  // 4 x 2
  const int m0 = blockIdx.y * BM, n0 = blockIdx.x * BN;
  const int wm = wrow * 32, wn = wcol * 32;
  const int lmod = lane & 15, lhalf = lane >> 4;
  const int kb8 = lhalf * 8;

  v8f accg[2][2], accu[2][2];
#pragma unroll
  for (int i = 0; i < 2; ++i)
#pragma unroll
    for (int j = 0; j < 2; ++j) { accg[i][j] = zero8(); accu[i][j] = zero8(); }

  for (int kb = 0; kb < K; kb += BK) {
    __syncthreads();
#pragma unroll
    for (int it = 0; it < 2; ++it) {
      int c = tid + it * 256;
      int r = c >> 2, o = (c & 3) * 8;
      *(v8bf*)&As[r * LDT + o] = *(const v8bf*)&A[(size_t)(m0 + r) * K + kb + o];
    }
    {
      int c = tid;                    // 256 chunks exactly
      int kk = c >> 3, nn = (c & 7) * 8;
      v8bf tg = *(const v8bf*)&B[(size_t)(kb + kk) * NTOT + n0 + nn];
      v8bf tu = *(const v8bf*)&B[(size_t)(kb + kk) * NTOT + kFFN + n0 + nn];
#pragma unroll
      for (int e = 0; e < 8; ++e) {
        Bg[(nn + e) * LDT + kk] = tg[e];
        Bu[(nn + e) * LDT + kk] = tu[e];
      }
    }
    __syncthreads();

    v16bf af[2], bg[2], bu[2];
#pragma unroll
    for (int i = 0; i < 2; ++i)
      af[i] = load_frag(&As[(wm + i * 16 + lmod) * LDT + kb8]);
#pragma unroll
    for (int j = 0; j < 2; ++j) {
      bg[j] = load_frag(&Bg[(wn + j * 16 + lmod) * LDT + kb8]);
      bu[j] = load_frag(&Bu[(wn + j * 16 + lmod) * LDT + kb8]);
    }
#pragma unroll
    for (int i = 0; i < 2; ++i)
#pragma unroll
      for (int j = 0; j < 2; ++j) {
        accg[i][j] = WMMA_BF16(af[i], bg[j], accg[i][j]);
        accu[i][j] = WMMA_BF16(af[i], bu[j], accu[i][j]);
      }
  }

#pragma unroll
  for (int i = 0; i < 2; ++i)
#pragma unroll
    for (int j = 0; j < 2; ++j) {
      int col = n0 + wn + j * 16 + lmod;
#pragma unroll
      for (int r = 0; r < 8; ++r) {
        int row = m0 + wm + i * 16 + r + 8 * lhalf;
        float g = accg[i][j][r], u = accu[i][j][r];
        float s = g / (1.0f + __expf(-g));   // silu
        act[(size_t)row * kFFN + col] = (bf16)(s * u);
      }
    }
}

// ---------------------------------------------------------------- WMMA flash attention
// grid (S/128, NH, B); 8 waves, wave w owns queries [q0+16w, q0+16w+15].
__global__ __launch_bounds__(256) void attn_kernel(
    const bf16* __restrict__ Q, const bf16* __restrict__ Kh,
    const bf16* __restrict__ Vh, bf16* __restrict__ ctx) {
  const int q0 = blockIdx.x * 128;
  const int h = blockIdx.y, b = blockIdx.z;
  const int kv = h / kG;
  const int wave = threadIdx.x >> 5, lane = threadIdx.x & 31;
  const int lmod = lane & 15, lhalf = lane >> 4;
  const int kb8 = lhalf * 8;

  const bf16* Qb = Q + ((size_t)(b * kNH + h) * kS + q0 + wave * 16) * kHD;
  const bf16* Kb = Kh + ((size_t)(b * kNKV + kv) * kS) * kHD;
  const bf16* Vb = Vh + ((size_t)(b * kNKV + kv) * kS) * kHD;

  __shared__ __align__(16) bf16 Ks[32 * 136];    // [key][hd], padded
  __shared__ __align__(16) bf16 VsT[128 * 40];   // [hd][key], padded
  __shared__ __align__(16) bf16 Ps[8][16 * 32];  // per-wave P tile

  // Q fragments: 4 chunks of 32 along head-dim
  v16bf aq[4];
#pragma unroll
  for (int c = 0; c < 4; ++c)
    aq[c] = load_frag(&Qb[(size_t)lmod * kHD + c * 32 + kb8]);

  float mi[8], li[8];
  v8f o[8];
#pragma unroll
  for (int r = 0; r < 8; ++r) { mi[r] = -1e30f; li[r] = 0.0f; }
#pragma unroll
  for (int t = 0; t < 8; ++t) o[t] = zero8();

  const int qmax_wave = q0 + wave * 16 + 15;
  const int kend = q0 + 128;

  for (int kb0 = 0; kb0 < kend; kb0 += 32) {
    __syncthreads();
    // stage K (row-major) and V (transposed) cooperatively
#pragma unroll
    for (int it = 0; it < 2; ++it) {
      int c = threadIdx.x + it * 256;
      int r = c >> 4, off = (c & 15) * 8;
      *(v8bf*)&Ks[r * 136 + off] = *(const v8bf*)&Kb[(size_t)(kb0 + r) * kHD + off];
      v8bf tv = *(const v8bf*)&Vb[(size_t)(kb0 + r) * kHD + off];
#pragma unroll
      for (int e = 0; e < 8; ++e) VsT[(off + e) * 40 + r] = tv[e];
    }
    __syncthreads();

    if (kb0 <= qmax_wave) {
      // scores for two 16-key subtiles
      v8f sc[2];
#pragma unroll
      for (int j = 0; j < 2; ++j) {
        sc[j] = zero8();
#pragma unroll
        for (int c = 0; c < 4; ++c) {
          v16bf bk = load_frag(&Ks[(j * 16 + lmod) * 136 + c * 32 + kb8]);
          sc[j] = WMMA_BF16(aq[c], bk, sc[j]);
        }
      }
      const int kcol0 = kb0 + lmod, kcol1 = kb0 + 16 + lmod;
      bf16* prow = Ps[wave];
#pragma unroll
      for (int r = 0; r < 8; ++r) {
        int qrow = q0 + wave * 16 + r + 8 * lhalf;
        float v0 = sc[0][r] * kSCALE; if (kcol0 > qrow) v0 = -1e9f;
        float v1 = sc[1][r] * kSCALE; if (kcol1 > qrow) v1 = -1e9f;
        float mx = rmax16(fmaxf(v0, v1));
        float nm = fmaxf(mi[r], mx);
        float p0 = __expf(v0 - nm), p1 = __expf(v1 - nm);
        float ps = rsum16(p0 + p1);
        float corr = __expf(mi[r] - nm);
        li[r] = li[r] * corr + ps;
        mi[r] = nm;
#pragma unroll
        for (int t = 0; t < 8; ++t) o[t][r] *= corr;
        int m = r + 8 * lhalf;
        prow[m * 32 + lmod] = (bf16)p0;
        prow[m * 32 + 16 + lmod] = (bf16)p1;
      }
      // P (16x32) x V (32x128)
      v16bf ap = load_frag(&prow[lmod * 32 + kb8]);
#pragma unroll
      for (int t = 0; t < 8; ++t) {
        v16bf bv = load_frag(&VsT[(t * 16 + lmod) * 40 + kb8]);
        o[t] = WMMA_BF16(ap, bv, o[t]);
      }
    }
  }

  // normalize + write ctx[b, q, h*HD + d]
#pragma unroll
  for (int t = 0; t < 8; ++t) {
#pragma unroll
    for (int r = 0; r < 8; ++r) {
      int qrow = q0 + wave * 16 + r + 8 * lhalf;
      float v = o[t][r] / li[r];
      ctx[((size_t)(b * kS + qrow)) * (size_t)(kNH * kHD) + h * kHD + t * 16 + lmod] = (bf16)v;
    }
  }
}

// ---------------------------------------------------------------- host launcher
extern "C" void kernel_launch(void* const* d_in, const int* in_sizes, int n_in,
                              void* d_out, int out_size, void* d_ws, size_t ws_size,
                              hipStream_t stream) {
  (void)in_sizes; (void)n_in; (void)out_size; (void)ws_size;
  const float* hidden  = (const float*)d_in[0];
  const int*   pos     = (const int*)d_in[1];
  const float* ln1_w   = (const float*)d_in[2];
  const float* w_qkv   = (const float*)d_in[3];
  const float* b_qkv   = (const float*)d_in[4];
  const float* w_dense = (const float*)d_in[5];
  const float* ln2_w   = (const float*)d_in[6];
  const float* w_h4h   = (const float*)d_in[7];
  const float* w_4hh   = (const float*)d_in[8];
  float* out = (float*)d_out;

  char* ws = (char*)d_ws;
  size_t off = 0;
  auto take = [&](size_t bytes) -> char* {
    char* p = ws + off;
    off += (bytes + 255) & ~(size_t)255;
    return p;
  };
  bf16*  ln1_bf    = (bf16*) take((size_t)kM * kH * 2);
  bf16*  wqkv_bf   = (bf16*) take((size_t)kH * kNQKV * 2);
  float* qkv_f     = (float*)take((size_t)kM * kNQKV * 4);
  bf16*  q_bf      = (bf16*) take((size_t)kB * kNH * kS * kHD * 2);
  bf16*  k_bf      = (bf16*) take((size_t)kB * kNKV * kS * kHD * 2);
  bf16*  v_bf      = (bf16*) take((size_t)kB * kNKV * kS * kHD * 2);
  bf16*  ctx_bf    = (bf16*) take((size_t)kM * kH * 2);
  bf16*  wdense_bf = (bf16*) take((size_t)kH * kH * 2);
  float* ln_in     = (float*)take((size_t)kM * kH * 4);
  bf16*  ln2_bf    = (bf16*) take((size_t)kM * kH * 2);
  bf16*  wh4h_bf   = (bf16*) take((size_t)kH * 2 * kFFN * 2);
  bf16*  act_bf    = (bf16*) take((size_t)kM * kFFN * 2);
  bf16*  w4hh_bf   = (bf16*) take((size_t)kFFN * kH * 2);

  // 1) weights -> bf16
  cast_f32_bf16<<<2048, 256, 0, stream>>>(w_qkv,   wqkv_bf,   (size_t)kH * kNQKV);
  cast_f32_bf16<<<2048, 256, 0, stream>>>(w_dense, wdense_bf, (size_t)kH * kH);
  cast_f32_bf16<<<4096, 256, 0, stream>>>(w_h4h,   wh4h_bf,   (size_t)kH * 2 * kFFN);
  cast_f32_bf16<<<4096, 256, 0, stream>>>(w_4hh,   w4hh_bf,   (size_t)kFFN * kH);

  // 2) rmsnorm1 -> bf16
  rmsnorm_kernel<<<kM, 256, 0, stream>>>(hidden, ln1_w, ln1_bf, kH);

  // 3) qkv = ln1 @ Wqkv + b  (f32 out)
  gemm_bf16<false, true, false><<<dim3(kNQKV / 128, kM / 128), 256, 0, stream>>>(
      ln1_bf, wqkv_bf, b_qkv, nullptr, qkv_f, kM, kNQKV, kH);

  // 4) rope + head-major split
  rope_kernel<<<kM, 128, 0, stream>>>(qkv_f, pos, q_bf, k_bf, v_bf);

  // 5) attention -> ctx (bf16)
  attn_kernel<<<dim3(kS / 128, kNH, kB), 256, 0, stream>>>(q_bf, k_bf, v_bf, ctx_bf);

  // 6) ln_in = ctx @ Wdense + hidden  (f32)
  gemm_bf16<false, false, true><<<dim3(kH / 128, kM / 128), 256, 0, stream>>>(
      ctx_bf, wdense_bf, nullptr, hidden, ln_in, kM, kH, kH);

  // 7) rmsnorm2 -> bf16
  rmsnorm_kernel<<<kM, 256, 0, stream>>>(ln_in, ln2_w, ln2_bf, kH);

  // 8) act = silu(ln2@Wg) * (ln2@Wu)  (bf16)
  gemm_gateup<<<dim3(kFFN / 64, kM / 128), 256, 0, stream>>>(ln2_bf, wh4h_bf, act_bf, kM, kH);

  // 9) out = act @ W4hh + ln_in  (f32)
  gemm_bf16<false, false, true><<<dim3(kH / 128, kM / 128), 256, 0, stream>>>(
      act_bf, w4hh_bf, nullptr, ln_in, out, kM, kH, kFFN);
}
